// TemporalConsistencyModule_21861383536793
// MI455X (gfx1250) — matrix operand order repaired and checked by reading, not compiled, not run on hardware
//
#include <hip/hip_runtime.h>
#include <hip/hip_bf16.h>
#include <math.h>

#define BB 4
#define TT 4096
#define DD 1024
#define NN 64
#define HH 8
#define HD 128
#define BT (BB*TT)

typedef __attribute__((ext_vector_type(16))) __bf16          v16bf;
typedef __attribute__((ext_vector_type(8)))  float           v8f;
typedef __attribute__((ext_vector_type(8)))  unsigned short  v8us;
typedef __attribute__((ext_vector_type(16))) unsigned short  v16us;

__device__ __forceinline__ unsigned short f2bf(float f) {
  unsigned int u = __float_as_uint(f);
  unsigned int r = u + 0x7FFFu + ((u >> 16) & 1u);   // round-to-nearest-even
  return (unsigned short)(r >> 16);
}

// Async global -> LDS copy of one 16-byte chunk (ASYNCcnt-tracked, no VGPR data)
__device__ __forceinline__ void async_b128(unsigned lds_off, const unsigned short* g) {
  asm volatile("global_load_async_to_lds_b128 %0, %1, off"
               :: "v"(lds_off), "v"(g) : "memory");
}
__device__ __forceinline__ void wait_async0() {
  asm volatile("s_wait_asynccnt 0x0" ::: "memory");
}

// ---------------------------------------------------------------------------
// bf16 WMMA GEMM:  C[M,N] = A[M,K] @ W[N,K]^T (+bias, + mode-specific epilogue)
// MODE 0: outf = acc+bias              (q / kv projections)
// MODE 1: outf = acc+bias+res          (attention out-proj + residual)
// MODE 2: outb = bf16(gelu(acc+bias))  (FFN up-proj)
// MODE 3: outf += acc+bias             (FFN down-proj accumulated into x)
// Block: 128 thr = 4 waves; block tile 128x64; wave tile 32x64 (8 accs).
// B staged in LDS as 64(N)x64(K) tiles, double-buffered, filled with
// GLOBAL_LOAD_ASYNC_TO_LDS_B128; 16 WMMAs per wave between barriers.
// ---------------------------------------------------------------------------
template<int MODE>
__global__ __launch_bounds__(128, 2)
void gemm_bf16_wmma(const unsigned short* __restrict__ A,
                    const unsigned short* __restrict__ W,
                    const float* __restrict__ bias,
                    const float* __restrict__ res,
                    float* __restrict__ outf,
                    unsigned short* __restrict__ outb,
                    int M, int N, int K)
{
  __shared__ __align__(16) unsigned short Bs[2][64 * 64];  // 2 x 8KB K-tiles

  const int lane = threadIdx.x & 31;
  const int wid  = threadIdx.x >> 5;
  const int hl   = lane >> 4;      // lane half (0/1)
  const int l16  = lane & 15;
  const int bN   = blockIdx.x * 64;
  const int bM   = blockIdx.y * 128;

  const unsigned short* arow0 = A + (size_t)(bM + wid * 32 + l16) * K;  // rows +0..15
  const unsigned short* arow1 = arow0 + (size_t)16 * K;                 // rows +16..31

  // Per-thread async-fill mapping (hoisted): col = t>>1, K-half = (t&1)*32
  const unsigned short* gfill =
      W + (size_t)(bN + (threadIdx.x >> 1)) * K + (threadIdx.x & 1) * 32;
  const unsigned lfill =
      (unsigned)(size_t)&Bs[0][(threadIdx.x >> 1) * 64 + (threadIdx.x & 1) * 32];

  v8f acc0[4] = {};
  v8f acc1[4] = {};

  // Prologue: async-fill buffer 0 with K-tile [0,64)
#pragma unroll
  for (int j = 0; j < 4; ++j) async_b128(lfill + j * 16u, gfill + j * 8);
  wait_async0();
  __syncthreads();

  int cur = 0;
  for (int kt = 0; kt < K; kt += 64) {
    // Overlap: kick off next K-tile's async fill while computing current one
    if (kt + 64 < K) {
      unsigned lnext = lfill + (unsigned)(cur ^ 1) * (64u * 64u * 2u);
      const unsigned short* gnext = gfill + kt + 64;
#pragma unroll
      for (int j = 0; j < 4; ++j) async_b128(lnext + j * 16u, gnext + j * 8);
      __builtin_prefetch(arow0 + kt + 128, 0, 3);   // A stream, 2 tiles ahead
      __builtin_prefetch(arow1 + kt + 128, 0, 3);
    }

#pragma unroll
    for (int kk = 0; kk < 64; kk += 32) {
      const int k = kt + kk;
      // A fragments per ISA 16-bit 16x32 layout (two contiguous 16B chunks/lane)
      v8us lo0 = *(const v8us*)(arow0 + k + hl * 8);
      v8us hi0 = *(const v8us*)(arow0 + k + 16 + hl * 8);
      v8us lo1 = *(const v8us*)(arow1 + k + hl * 8);
      v8us hi1 = *(const v8us*)(arow1 + k + 16 + hl * 8);
      v16us au0, au1;
#pragma unroll
      for (int i = 0; i < 8; ++i) {
        au0[i] = lo0[i]; au0[i + 8] = hi0[i];
        au1[i] = lo1[i]; au1[i + 8] = hi1[i];
      }
      v16bf a0 = __builtin_bit_cast(v16bf, au0);
      v16bf a1 = __builtin_bit_cast(v16bf, au1);

      // Preload all 4 B fragments (8 ds_load_b128 as one clause, single wait)
      v16bf bfr[4];
#pragma unroll
      for (int nt = 0; nt < 4; ++nt)
        bfr[nt] = __builtin_bit_cast(v16bf, __builtin_shufflevector(
            *(const v8us*)&Bs[cur][(nt * 16 + l16) * 64 + kk + hl * 8],
            *(const v8us*)&Bs[cur][(nt * 16 + l16) * 64 + kk + 16 + hl * 8],
            0,1,2,3,4,5,6,7,8,9,10,11,12,13,14,15));

      // 8 back-to-back WMMAs, no intermediate memory waits
#pragma unroll
      for (int nt = 0; nt < 4; ++nt) {
        acc0[nt] = __builtin_amdgcn_wmma_f32_16x16x32_bf16(
            false, a0, false, bfr[nt], (short)0, acc0[nt], false, false);
        acc1[nt] = __builtin_amdgcn_wmma_f32_16x16x32_bf16(
            false, a1, false, bfr[nt], (short)0, acc1[nt], false, false);
      }
    }

    wait_async0();       // next buffer resident (usually already done)
    __syncthreads();     // all waves done reading Bs[cur] before it is refilled
    cur ^= 1;
  }

  // Epilogue: C/D layout VGPR r -> (M = r + 8*half, N = l16)
#pragma unroll
  for (int nt = 0; nt < 4; ++nt) {
#pragma unroll
    for (int r = 0; r < 8; ++r) {
      int col  = bN + nt * 16 + l16;
      int row0 = bM + wid * 32 + hl * 8 + r;
      int row1 = row0 + 16;
      float v0 = acc0[nt][r] + bias[col];
      float v1 = acc1[nt][r] + bias[col];
      size_t o0 = (size_t)row0 * N + col;
      size_t o1 = (size_t)row1 * N + col;
      if (MODE == 0)      { outf[o0] = v0; outf[o1] = v1; }
      else if (MODE == 1) { outf[o0] = v0 + res[o0]; outf[o1] = v1 + res[o1]; }
      else if (MODE == 2) {                       // exact GELU
        float g0 = 0.5f * v0 * (1.0f + erff(v0 * 0.70710678118654752f));
        float g1 = 0.5f * v1 * (1.0f + erff(v1 * 0.70710678118654752f));
        outb[o0] = f2bf(g0); outb[o1] = f2bf(g1);
      } else              { outf[o0] += v0; outf[o1] += v1; }
    }
  }
}

// ---------------------------------------------------------------------------
// fp32 -> bf16 conversion (grid-stride)
// ---------------------------------------------------------------------------
__global__ void cvt_kernel(const float* __restrict__ in,
                           unsigned short* __restrict__ out, size_t n)
{
  size_t i = (size_t)blockIdx.x * blockDim.x + threadIdx.x;
  size_t stride = (size_t)gridDim.x * blockDim.x;
  for (; i < n; i += stride) out[i] = f2bf(in[i]);
}

__global__ void zero_kernel(float* p, int n)
{
  int i = blockIdx.x * blockDim.x + threadIdx.x;
  if (i < n) p[i] = 0.0f;
}

__global__ void copyf_kernel(const float* __restrict__ in,
                             float* __restrict__ out, size_t n)
{
  size_t i = (size_t)blockIdx.x * blockDim.x + threadIdx.x;
  size_t stride = (size_t)gridDim.x * blockDim.x;
  for (; i < n; i += stride) out[i] = in[i];
}

// ---------------------------------------------------------------------------
// Cross-attention: one wave per (b,h,t) row.  N=64 keys, hd=128.
// Each lane owns keys {lane, lane+32}; softmax via shfl_xor; ctx: 4 dims/lane.
// Probabilities also accumulated into accbn[b][n] for the top-k ranking.
// ---------------------------------------------------------------------------
__global__ __launch_bounds__(256)
void attn_kernel(const float* __restrict__ qf, const float* __restrict__ kvf,
                 float* __restrict__ accbn, unsigned short* __restrict__ ctxb)
{
  const int lane = threadIdx.x & 31;
  const int wid  = threadIdx.x >> 5;
  int row = blockIdx.x * 8 + wid;              // 0 .. B*H*T-1
  int t = row % TT;
  int h = (row / TT) % HH;
  int b = row / (TT * HH);

  const float* q  = qf  + ((size_t)b * TT + t) * DD + h * HD;
  const float* kb = kvf + (size_t)b * NN * (2 * DD) + h * HD;
  const float* k0 = kb + (size_t)lane * (2 * DD);
  const float* k1 = kb + (size_t)(lane + 32) * (2 * DD);

  float s0 = 0.f, s1 = 0.f;
#pragma unroll 4
  for (int d = 0; d < HD; ++d) { float qd = q[d]; s0 += qd * k0[d]; s1 += qd * k1[d]; }
  const float scale = 0.0883883476483184f;     // 1/sqrt(128)
  s0 *= scale; s1 *= scale;

  float m = fmaxf(s0, s1);
#pragma unroll
  for (int off = 16; off > 0; off >>= 1) m = fmaxf(m, __shfl_xor(m, off, 32));
  float e0 = __expf(s0 - m), e1 = __expf(s1 - m);
  float s = e0 + e1;
#pragma unroll
  for (int off = 16; off > 0; off >>= 1) s += __shfl_xor(s, off, 32);
  float inv = 1.0f / s;
  float p0 = e0 * inv, p1 = e1 * inv;

  atomicAdd(&accbn[b * NN + lane], p0);
  atomicAdd(&accbn[b * NN + lane + 32], p1);

  const float* vb = kvf + (size_t)b * NN * (2 * DD) + DD + h * HD;
  int d0 = lane * 4;
  float c0 = 0.f, c1 = 0.f, c2 = 0.f, c3 = 0.f;
  for (int n = 0; n < NN; ++n) {
    float pn = __shfl((n < 32) ? p0 : p1, n & 31, 32);
    const float* vr = vb + (size_t)n * (2 * DD) + d0;
    c0 += pn * vr[0]; c1 += pn * vr[1]; c2 += pn * vr[2]; c3 += pn * vr[3];
  }
  unsigned short* o = ctxb + ((size_t)b * TT + t) * DD + h * HD + d0;
  o[0] = f2bf(c0); o[1] = f2bf(c1); o[2] = f2bf(c2); o[3] = f2bf(c3);
}

// ---------------------------------------------------------------------------
// LayerNorm: one wave per row of 1024; outputs bf16 (next GEMM's A operand)
// ---------------------------------------------------------------------------
__global__ __launch_bounds__(256)
void ln_kernel(const float* __restrict__ x, const float* __restrict__ gamma,
               const float* __restrict__ beta, unsigned short* __restrict__ out)
{
  const int lane = threadIdx.x & 31;
  const int wid  = threadIdx.x >> 5;
  int row = blockIdx.x * 8 + wid;
  const float* xr = x + (size_t)row * DD;
  float s = 0.f, ss = 0.f;
#pragma unroll
  for (int i = 0; i < DD / 32; ++i) { float v = xr[lane + i * 32]; s += v; ss += v * v; }
#pragma unroll
  for (int off = 16; off > 0; off >>= 1) {
    s  += __shfl_xor(s,  off, 32);
    ss += __shfl_xor(ss, off, 32);
  }
  float mu  = s * (1.0f / DD);
  float var = ss * (1.0f / DD) - mu * mu;
  float inv = rsqrtf(var + 1e-5f);
  unsigned short* orow = out + (size_t)row * DD;
#pragma unroll
  for (int i = 0; i < DD / 32; ++i) {
    int cidx = lane + i * 32;
    float v = (xr[cidx] - mu) * inv * gamma[cidx] + beta[cidx];
    orow[cidx] = f2bf(v);
  }
}

// ---------------------------------------------------------------------------
// Column mean over R rows: out[b*D+d] = mean_r in[(b*R+r)*D+d]   (B*D threads)
// ---------------------------------------------------------------------------
__global__ void colmean_kernel(const float* __restrict__ in,
                               float* __restrict__ out, int R)
{
  int id = blockIdx.x * blockDim.x + threadIdx.x;
  int b = id / DD, d = id % DD;
  const float* p = in + (size_t)b * R * DD + d;
  float s = 0.f;
  for (int r = 0; r < R; ++r) s += p[(size_t)r * DD];
  out[id] = s / (float)R;
}

// gate = sigmoid([fs, ms] @ gw^T + gb); nmem = g*fs + (1-g)*ms  (B*D threads)
__global__ void gate_kernel(const float* __restrict__ fs, const float* __restrict__ ms,
                            const float* __restrict__ gw, const float* __restrict__ gb,
                            float* __restrict__ nmem)
{
  int id = blockIdx.x * blockDim.x + threadIdx.x;
  int b = id / DD, d = id % DD;
  const float* wr = gw + (size_t)d * (2 * DD);
  const float* f  = fs + (size_t)b * DD;
  const float* mm = ms + (size_t)b * DD;
  float z = gb[d];
  for (int k = 0; k < DD; ++k) z += f[k] * wr[k];
  for (int k = 0; k < DD; ++k) z += mm[k] * wr[DD + k];
  float g = 1.0f / (1.0f + __expf(-z));
  nmem[id] = g * f[d] + (1.0f - g) * mm[d];
}

__global__ void topk_kernel(const float* __restrict__ accbn, int* __restrict__ idx)
{
  int b = threadIdx.x;
  if (b >= BB) return;
  float v[NN];
  for (int n = 0; n < NN; ++n) v[n] = accbn[b * NN + n];
  for (int j = 0; j < 3; ++j) {
    float best = -1e30f; int bi = 0;
    for (int n = 0; n < NN; ++n)
      if (v[n] > best) { best = v[n]; bi = n; }
    idx[b * 3 + j] = bi;
    v[bi] = -1e30f;
  }
}

__global__ void memupd_kernel(const float* __restrict__ mem_in,
                              const float* __restrict__ nmem,
                              const int* __restrict__ idx,
                              float* __restrict__ mem_out)
{
  int b = blockIdx.x / 3, j = blockIdx.x % 3;
  int row = idx[b * 3 + j];
  size_t base = ((size_t)b * NN + row) * DD;
  for (int d = threadIdx.x; d < DD; d += blockDim.x)
    mem_out[base + d] = 0.7f * mem_in[base + d] + 0.3f * nmem[(size_t)b * DD + d];
}

// ---------------------------------------------------------------------------
extern "C" void kernel_launch(void* const* d_in, const int* in_sizes, int n_in,
                              void* d_out, int out_size, void* d_ws, size_t ws_size,
                              hipStream_t stream)
{
  const float* cf   = (const float*)d_in[0];
  const float* mem  = (const float*)d_in[1];
  const float* ipw  = (const float*)d_in[2];
  const float* ipb  = (const float*)d_in[3];
  const float* opw  = (const float*)d_in[4];
  const float* opb  = (const float*)d_in[5];
  const float* lng  = (const float*)d_in[6];
  const float* lnbt = (const float*)d_in[7];
  const float* w1   = (const float*)d_in[8];
  const float* b1   = (const float*)d_in[9];
  const float* w2   = (const float*)d_in[10];
  const float* b2   = (const float*)d_in[11];
  const float* gw   = (const float*)d_in[12];
  const float* gb   = (const float*)d_in[13];

  float* xout   = (float*)d_out;                 // output 0: x [B,T,D]
  float* memout = xout + (size_t)BT * DD;        // output 1: entity_memory [B,N,D]

  char* wsp = (char*)d_ws;
  auto alloc = [&](size_t bytes) {
    char* p = wsp; wsp += (bytes + 255) & ~(size_t)255; return p;
  };
  unsigned short* xb    = (unsigned short*)alloc((size_t)BT * DD * 2);
  unsigned short* memb  = (unsigned short*)alloc((size_t)BB * NN * DD * 2);
  unsigned short* wqkvb = (unsigned short*)alloc((size_t)3 * DD * DD * 2);
  unsigned short* woutb = (unsigned short*)alloc((size_t)DD * DD * 2);
  unsigned short* w1b   = (unsigned short*)alloc((size_t)3 * 2 * DD * DD * 2);
  unsigned short* w2b   = (unsigned short*)alloc((size_t)3 * 2 * DD * DD * 2);
  unsigned short* ctxb  = (unsigned short*)alloc((size_t)BT * DD * 2);
  unsigned short* lnxb  = (unsigned short*)alloc((size_t)BT * DD * 2);
  unsigned short* hb    = (unsigned short*)alloc((size_t)BT * 2 * DD * 2);
  float* qf    = (float*)alloc((size_t)BT * DD * 4);
  float* kvf   = (float*)alloc((size_t)BB * NN * 2 * DD * 4);
  float* accbn = (float*)alloc((size_t)BB * NN * 4);
  float* fsum  = (float*)alloc((size_t)BB * DD * 4);
  float* msum  = (float*)alloc((size_t)BB * DD * 4);
  float* nmem  = (float*)alloc((size_t)BB * DD * 4);
  int*   tki   = (int*)alloc((size_t)BB * 3 * 4);

  dim3 blk(256);
  dim3 gblk(128);   // GEMM uses 4 waves/block

  // fp32 -> bf16 operand staging
  cvt_kernel<<<4096, blk, 0, stream>>>(cf,  xb,    (size_t)BT * DD);
  cvt_kernel<<<256,  blk, 0, stream>>>(mem, memb,  (size_t)BB * NN * DD);
  cvt_kernel<<<2048, blk, 0, stream>>>(ipw, wqkvb, (size_t)3 * DD * DD);
  cvt_kernel<<<1024, blk, 0, stream>>>(opw, woutb, (size_t)DD * DD);
  cvt_kernel<<<2048, blk, 0, stream>>>(w1,  w1b,   (size_t)3 * 2 * DD * DD);
  cvt_kernel<<<2048, blk, 0, stream>>>(w2,  w2b,   (size_t)3 * 2 * DD * DD);

  // q = x @ Wq^T + bq      [16384 x 1024]
  gemm_bf16_wmma<0><<<dim3(DD / 64, BT / 128), gblk, 0, stream>>>(
      xb, wqkvb, ipb, nullptr, qf, nullptr, BT, DD, DD);
  // [k|v] = mem @ [Wk|Wv]^T + [bk|bv]    [256 x 2048]
  gemm_bf16_wmma<0><<<dim3(2 * DD / 64, (BB * NN) / 128), gblk, 0, stream>>>(
      memb, wqkvb + (size_t)DD * DD, ipb + DD, nullptr, kvf, nullptr,
      BB * NN, 2 * DD, DD);

  zero_kernel<<<1, 256, 0, stream>>>(accbn, BB * NN);
  attn_kernel<<<BB * HH * TT / 8, blk, 0, stream>>>(qf, kvf, accbn, ctxb);

  // x = current + ctx @ Wo^T + bo
  gemm_bf16_wmma<1><<<dim3(DD / 64, BT / 128), gblk, 0, stream>>>(
      ctxb, woutb, opb, cf, xout, nullptr, BT, DD, DD);

  // 3 pre-LN FFN layers
  for (int i = 0; i < 3; ++i) {
    ln_kernel<<<BT / 8, blk, 0, stream>>>(xout, lng + (size_t)i * DD,
                                          lnbt + (size_t)i * DD, lnxb);
    gemm_bf16_wmma<2><<<dim3(2 * DD / 64, BT / 128), gblk, 0, stream>>>(
        lnxb, w1b + (size_t)i * 2 * DD * DD, b1 + (size_t)i * 2 * DD,
        nullptr, nullptr, hb, BT, 2 * DD, DD);
    gemm_bf16_wmma<3><<<dim3(DD / 64, BT / 128), gblk, 0, stream>>>(
        hb, w2b + (size_t)i * 2 * DD * DD, b2 + (size_t)i * DD,
        nullptr, xout, nullptr, BT, DD, 2 * DD);
  }

  // gated memory update
  colmean_kernel<<<(BB * DD) / 256, blk, 0, stream>>>(cf, fsum, TT);
  colmean_kernel<<<(BB * DD) / 256, blk, 0, stream>>>(mem, msum, NN);
  gate_kernel<<<(BB * DD) / 256, blk, 0, stream>>>(fsum, msum, gw, gb, nmem);
  topk_kernel<<<1, 32, 0, stream>>>(accbn, tki);
  copyf_kernel<<<256, blk, 0, stream>>>(mem, memout, (size_t)BB * NN * DD);
  memupd_kernel<<<BB * 3, blk, 0, stream>>>(mem, nmem, tki, memout);
}
